// GraphCCS_23167053594654
// MI455X (gfx1250) — matrix operand order repaired and checked by reading, not compiled, not run on hardware
//
#include <hip/hip_runtime.h>
#include <math.h>

#define NN 200000
#define NE 800000
#define NG 8192
#define HH 200
#define KP_H 224    // 200 padded to mult of 32
#define NQ_H 208    // 200 padded to mult of 16
#define KP_NODE 96  // 74  -> 96
#define KP_EDGE 32  // 12  -> 32
#define NQ_G 608    // 600 -> 608
#define NQ_O1 1024

typedef _Float16 v16h __attribute__((ext_vector_type(16)));
typedef _Float16 h8   __attribute__((ext_vector_type(8)));
typedef float    v8f  __attribute__((ext_vector_type(8)));

#define MODE_NONE   0
#define MODE_RELU   1
#define MODE_LS_RES 2

// ---------------- helpers ----------------
__device__ __forceinline__ void atomicMaxF(float* addr, float val) {
    if (val >= 0.0f) atomicMax((int*)addr, __float_as_int(val));
    else             atomicMin((unsigned int*)addr, __float_as_uint(val));
}

// ---------------- generic fills / converts ----------------
__global__ void k_fill(float* __restrict__ p, float v, size_t n) {
    size_t i = (size_t)blockIdx.x * blockDim.x + threadIdx.x;
    size_t stride = (size_t)gridDim.x * blockDim.x;
    for (; i < n; i += stride) p[i] = v;
}

// A [M,K] f32 -> [M,Kp] f16 (zero-padded K..Kp)
__global__ void k_cvt_a16(const float* __restrict__ in, _Float16* __restrict__ out,
                          int M, int K, int Kp) {
    size_t tot = (size_t)M * Kp;
    size_t i = (size_t)blockIdx.x * blockDim.x + threadIdx.x;
    size_t stride = (size_t)gridDim.x * blockDim.x;
    for (; i < tot; i += stride) {
        size_t m = i / (size_t)Kp;
        int k = (int)(i - m * (size_t)Kp);
        out[i] = (k < K) ? (_Float16)in[m * (size_t)K + k] : (_Float16)0.0f;
    }
}

// W [K,N] f32 row-major -> Bt [Nq,Kp] f16 (K-contiguous per output column)
__global__ void k_cvt_w16t(const float* __restrict__ W, _Float16* __restrict__ out,
                           int K, int N, int Kp, int Nq) {
    size_t tot = (size_t)Nq * Kp;
    size_t i = (size_t)blockIdx.x * blockDim.x + threadIdx.x;
    size_t stride = (size_t)gridDim.x * blockDim.x;
    for (; i < tot; i += stride) {
        int n = (int)(i / (size_t)Kp);
        int k = (int)(i - (size_t)n * Kp);
        out[i] = (n < N && k < K) ? (_Float16)W[(size_t)k * N + n] : (_Float16)0.0f;
    }
}

// ---------------- WMMA GEMM: C[M,Nout] = act(A16[M,Kp] @ Bt16[Nq,Kp]^T + bias) ----------------
// 256 thr = 8 waves; wave w owns 16-row output strip, iterates all 16-col tiles.
__global__ __launch_bounds__(256) void k_gemm_wmma(
    const _Float16* __restrict__ A, const _Float16* __restrict__ Bt,
    const float* __restrict__ bias, float* __restrict__ C,
    const float* __restrict__ ls, int M, int Nout, int Kp, int Nq, int mode)
{
    const int wave = threadIdx.x >> 5;
    const int lane = threadIdx.x & 31;
    const int half = lane >> 4;
    const int ln   = lane & 15;
    const long mt  = (long)blockIdx.x * 8 + wave;
    if (mt * 16 >= M) return;

    const int ktiles = Kp >> 5;
    const int ntiles = Nq >> 4;
    const _Float16* Arow = A + (size_t)(mt * 16 + ln) * Kp;

    for (int nt = 0; nt < ntiles; ++nt) {
        const int ncol = nt * 16 + ln;
        const float bv = (bias != nullptr && ncol < Nout) ? bias[ncol] : 0.0f;
        v8f acc;
#pragma unroll
        for (int r = 0; r < 8; ++r) acc[r] = bv;

        const _Float16* Bn = Bt + (size_t)(nt * 16 + ln) * Kp;
        for (int kt = 0; kt < ktiles; ++kt) {
            // A 16x32 fragment: lane row = ln; K = {half*8+e} U {16+half*8+e}
            const _Float16* ap = Arow + kt * 32 + half * 8;
            h8 a0 = *(const h8*)(ap);
            h8 a1 = *(const h8*)(ap + 16);
            v16h af = __builtin_shufflevector(a0, a1, 0,1,2,3,4,5,6,7,8,9,10,11,12,13,14,15);
            // B 32x16 fragment: lane col = ln; K = {half*8+e} U {16+half*8+e}
            const _Float16* bp = Bn + kt * 32 + half * 8;
            h8 b0 = *(const h8*)(bp);
            h8 b1 = *(const h8*)(bp + 16);
            v16h bf = __builtin_shufflevector(b0, b1, 0,1,2,3,4,5,6,7,8,9,10,11,12,13,14,15);
            acc = __builtin_amdgcn_wmma_f32_16x16x32_f16(false, af, false, bf,
                                                         (short)0, acc, false, false);
        }
        if (ncol < Nout) {
#pragma unroll
            for (int r = 0; r < 8; ++r) {
                long mrow = mt * 16 + half * 8 + r;
                float* cp = C + (size_t)mrow * Nout + ncol;
                float v = acc[r];
                if (mode == MODE_RELU)        v = v > 0.0f ? v : 0.0f;
                else if (mode == MODE_LS_RES) v = (v > 0.0f ? v : 0.0f) * ls[ncol] + *cp;
                *cp = v;
            }
        }
    }
}

// ---------------- edge softmax passes (one block per edge, f = tid) ----------------
__global__ void k_edge_max(const float* __restrict__ h, const float* __restrict__ e,
                           const int* __restrict__ src, const int* __restrict__ dst,
                           float* __restrict__ nmax) {
    int eid = blockIdx.x, f = threadIdx.x;
    if (f >= HH) return;
    int s = src[eid], d = dst[eid];
    float m = h[(size_t)s * HH + f] + __builtin_nontemporal_load(&e[(size_t)eid * HH + f]);
    atomicMaxF(&nmax[(size_t)d * HH + f], m);
}

__global__ void k_edge_expsum(const float* __restrict__ h, const float* __restrict__ e,
                              const int* __restrict__ src, const int* __restrict__ dst,
                              const float* __restrict__ nmax,
                              float* __restrict__ den, float* __restrict__ wsum) {
    int eid = blockIdx.x, f = threadIdx.x;
    if (f >= HH) return;
    int s = src[eid], d = dst[eid];
    float m  = h[(size_t)s * HH + f] + __builtin_nontemporal_load(&e[(size_t)eid * HH + f]);
    float ex = expf(m - nmax[(size_t)d * HH + f]);
    atomicAdd(&den[(size_t)d * HH + f], ex);
    atomicAdd(&wsum[(size_t)d * HH + f], m * ex);
}

// agg = wsum/den -> padded f16 GEMM operand [NN, KP_H]; block per node, 224 threads
__global__ void k_agg16(const float* __restrict__ wsum, const float* __restrict__ den,
                        _Float16* __restrict__ out) {
    int v = blockIdx.x, f = threadIdx.x;  // f in [0,224)
    float val = 0.0f;
    if (f < HH) {
        float d = den[(size_t)v * HH + f];
        val = (d > 0.0f) ? wsum[(size_t)v * HH + f] / d : 0.0f;
    }
    out[(size_t)v * KP_H + f] = (_Float16)val;
}

// ---------------- readout ----------------
__global__ void k_seg_g(const float* __restrict__ h, const int* __restrict__ n2g,
                        float* __restrict__ g) {
    int v = blockIdx.x, f = threadIdx.x;
    if (f >= HH) return;
    atomicAdd(&g[(size_t)n2g[v] * HH + f], h[(size_t)v * HH + f]);
}

// one wave32 per node: z = leaky_relu(dot(relu(g), Wl[0:H]) + dot(h, Wl[H:2H]) + b)
__global__ void k_node_logits(const float* __restrict__ g, const float* __restrict__ h,
                              const int* __restrict__ n2g, const float* __restrict__ Wl,
                              const float* __restrict__ bl, float* __restrict__ z,
                              float* __restrict__ gmax) {
    int wave = threadIdx.x >> 5, lane = threadIdx.x & 31;
    long v = (long)blockIdx.x * 8 + wave;
    if (v >= NN) return;
    int gid = n2g[v];
    const float* gr = g + (size_t)gid * HH;
    const float* hr = h + (size_t)v * HH;
    float acc = 0.0f;
    for (int f = lane; f < HH; f += 32)
        acc += fmaxf(gr[f], 0.0f) * Wl[f] + hr[f] * Wl[HH + f];
    for (int o = 16; o > 0; o >>= 1) acc += __shfl_down(acc, o, 32);
    if (lane == 0) {
        float zz = acc + bl[0];
        zz = zz > 0.0f ? zz : 0.01f * zz;
        z[v] = zz;
        atomicMaxF(&gmax[gid], zz);
    }
}

__global__ void k_node_expsum(const float* __restrict__ z, const int* __restrict__ n2g,
                              const float* __restrict__ gmax, float* __restrict__ gsum) {
    size_t v = (size_t)blockIdx.x * blockDim.x + threadIdx.x;
    if (v >= NN) return;
    int gid = n2g[v];
    atomicAdd(&gsum[gid], expf(z[v] - gmax[gid]));
}

__global__ void k_ctx_scatter(const float* __restrict__ hv, const float* __restrict__ z,
                              const int* __restrict__ n2g, const float* __restrict__ gmax,
                              const float* __restrict__ gsum, float* __restrict__ ctx) {
    int v = blockIdx.x, f = threadIdx.x;
    if (f >= HH) return;
    int gid = n2g[v];
    float a = expf(z[v] - gmax[gid]) / gsum[gid];
    atomicAdd(&ctx[(size_t)gid * HH + f], hv[(size_t)v * HH + f] * a);
}

__global__ void k_elu(float* __restrict__ p, size_t n) {
    size_t i = (size_t)blockIdx.x * blockDim.x + threadIdx.x;
    if (i >= n) return;
    float x = p[i];
    p[i] = x > 0.0f ? x : expf(x) - 1.0f;
}

__global__ void k_gru(const float* __restrict__ gi, const float* __restrict__ gh,
                      float* __restrict__ g) {
    size_t i = (size_t)blockIdx.x * blockDim.x + threadIdx.x;
    if (i >= (size_t)NG * HH) return;
    size_t gg = i / HH;
    int f = (int)(i - gg * HH);
    const float* gir = gi + gg * 600;
    const float* ghr = gh + gg * 600;
    float r = 1.0f / (1.0f + expf(-(gir[f]          + ghr[f])));
    float u = 1.0f / (1.0f + expf(-(gir[HH + f]     + ghr[HH + f])));
    float n = tanhf(gir[2 * HH + f] + r * ghr[2 * HH + f]);
    g[i] = (1.0f - u) * n + u * g[i];
}

__global__ void k_out2(const float* __restrict__ o1, const float* __restrict__ w2,
                       const float* __restrict__ b2, float* __restrict__ out) {
    __shared__ float red[256];
    int g = blockIdx.x, tid = threadIdx.x;
    float s = 0.0f;
    for (int j = tid; j < 1024; j += 256) s += o1[(size_t)g * 1024 + j] * w2[j];
    red[tid] = s; __syncthreads();
    for (int st = 128; st > 0; st >>= 1) { if (tid < st) red[tid] += red[tid + st]; __syncthreads(); }
    if (tid == 0) out[g] = red[0] + b2[0];
}

// ---------------- launch ----------------
extern "C" void kernel_launch(void* const* d_in, const int* in_sizes, int n_in,
                              void* d_out, int out_size, void* d_ws, size_t ws_size,
                              hipStream_t stream) {
    const float* node_feats = (const float*)d_in[0];
    const float* edge_feats = (const float*)d_in[1];
    const int*   src        = (const int*)d_in[2];
    const int*   dst        = (const int*)d_in[3];
    const int*   n2g        = (const int*)d_in[4];
    const float* atom_W = (const float*)d_in[5];
    const float* atom_b = (const float*)d_in[6];
    const float* bond_W = (const float*)d_in[7];
    const float* bond_b = (const float*)d_in[8];
    const float* mlp_W  = (const float*)d_in[9];
    const float* mlp_b  = (const float*)d_in[10];
    const float* ls     = (const float*)d_in[11];
    const float* logit_W = (const float*)d_in[12];
    const float* logit_b = (const float*)d_in[13];
    const float* proj_W  = (const float*)d_in[14];
    const float* proj_b  = (const float*)d_in[15];
    const float* gru_Wih = (const float*)d_in[16];
    const float* gru_Whh = (const float*)d_in[17];
    const float* gru_bih = (const float*)d_in[18];
    const float* gru_bhh = (const float*)d_in[19];
    const float* out1_W = (const float*)d_in[20];
    const float* out1_b = (const float*)d_in[21];
    const float* out2_W = (const float*)d_in[22];
    const float* out2_b = (const float*)d_in[23];
    float* out = (float*)d_out;

    // workspace layout (all counts multiple of 4 floats => 16B alignment for f16 vectors)
    float* f = (float*)d_ws;
    float* e_buf = f; f += (size_t)NE * HH;
    float* h_buf = f; f += (size_t)NN * HH;
    float* nmax  = f; f += (size_t)NN * HH;
    float* den   = f; f += (size_t)NN * HH;
    float* wsum  = f; f += (size_t)NN * HH;   // also hv
    float* gbuf  = f; f += (size_t)NG * HH;
    float* ctx   = f; f += (size_t)NG * HH;
    float* gi    = f; f += (size_t)NG * 600;
    float* gh    = f; f += (size_t)NG * 600;
    float* zbuf  = f; f += (size_t)NN;
    float* gmax  = f; f += (size_t)NG;
    float* gsum  = f; f += (size_t)NG;
    float* o1    = f; f += (size_t)NG * 1024;
    _Float16* x16 = (_Float16*)f;
    _Float16* w16 = x16 + (size_t)NN * KP_H;
    _Float16* c16 = w16 + (size_t)NQ_O1 * KP_H;
    _Float16* g16 = c16 + (size_t)NG * KP_H;

    auto cdiv = [](long a, long b) { return (unsigned)((a + b - 1) / b); };
    const size_t NH = (size_t)NN * HH;

    // ---- encoders ----
    k_cvt_a16<<<cdiv((long)NN * KP_NODE, 256), 256, 0, stream>>>(node_feats, x16, NN, 74, KP_NODE);
    k_cvt_w16t<<<cdiv((long)NQ_H * KP_NODE, 256), 256, 0, stream>>>(atom_W, w16, 74, HH, KP_NODE, NQ_H);
    k_gemm_wmma<<<cdiv(NN / 16, 8), 256, 0, stream>>>(x16, w16, atom_b, h_buf, nullptr,
                                                      NN, HH, KP_NODE, NQ_H, MODE_NONE);
    k_cvt_a16<<<cdiv((long)NE * KP_EDGE, 256), 256, 0, stream>>>(edge_feats, x16, NE, 12, KP_EDGE);
    k_cvt_w16t<<<cdiv((long)NQ_H * KP_EDGE, 256), 256, 0, stream>>>(bond_W, w16, 12, HH, KP_EDGE, NQ_H);
    k_gemm_wmma<<<cdiv(NE / 16, 8), 256, 0, stream>>>(x16, w16, bond_b, e_buf, nullptr,
                                                      NE, HH, KP_EDGE, NQ_H, MODE_NONE);

    // ---- message passing layers ----
    for (int i = 0; i < 5; ++i) {
        k_fill<<<8192, 256, 0, stream>>>(nmax, -__builtin_inff(), NH);
        k_fill<<<8192, 256, 0, stream>>>(den, 0.0f, NH);
        k_fill<<<8192, 256, 0, stream>>>(wsum, 0.0f, NH);
        k_edge_max<<<NE, 224, 0, stream>>>(h_buf, e_buf, src, dst, nmax);
        k_edge_expsum<<<NE, 224, 0, stream>>>(h_buf, e_buf, src, dst, nmax, den, wsum);
        k_agg16<<<NN, 224, 0, stream>>>(wsum, den, x16);
        k_cvt_w16t<<<cdiv((long)NQ_H * KP_H, 256), 256, 0, stream>>>(
            mlp_W + (size_t)i * HH * HH, w16, HH, HH, KP_H, NQ_H);
        k_gemm_wmma<<<cdiv(NN / 16, 8), 256, 0, stream>>>(x16, w16, mlp_b + (size_t)i * HH,
                                                          h_buf, ls + (size_t)i * HH,
                                                          NN, HH, KP_H, NQ_H, MODE_LS_RES);
    }

    // ---- AttentiveFP readout ----
    k_fill<<<256, 256, 0, stream>>>(gbuf, 0.0f, (size_t)NG * HH);
    k_seg_g<<<NN, 224, 0, stream>>>(h_buf, n2g, gbuf);
    k_cvt_a16<<<cdiv((long)NN * KP_H, 256), 256, 0, stream>>>(h_buf, x16, NN, HH, KP_H); // h fixed now

    for (int t = 0; t < 2; ++t) {
        k_fill<<<32, 256, 0, stream>>>(gmax, -__builtin_inff(), (size_t)NG);
        k_fill<<<32, 256, 0, stream>>>(gsum, 0.0f, (size_t)NG);
        k_fill<<<256, 256, 0, stream>>>(ctx, 0.0f, (size_t)NG * HH);
        k_node_logits<<<cdiv(NN, 8), 256, 0, stream>>>(gbuf, h_buf, n2g,
                                                       logit_W + (size_t)t * 2 * HH,
                                                       logit_b + t, zbuf, gmax);
        k_node_expsum<<<cdiv(NN, 256), 256, 0, stream>>>(zbuf, n2g, gmax, gsum);
        k_cvt_w16t<<<cdiv((long)NQ_H * KP_H, 256), 256, 0, stream>>>(
            proj_W + (size_t)t * HH * HH, w16, HH, HH, KP_H, NQ_H);
        k_gemm_wmma<<<cdiv(NN / 16, 8), 256, 0, stream>>>(x16, w16, proj_b + (size_t)t * HH,
                                                          wsum /*hv*/, nullptr,
                                                          NN, HH, KP_H, NQ_H, MODE_NONE);
        k_ctx_scatter<<<NN, 224, 0, stream>>>(wsum, zbuf, n2g, gmax, gsum, ctx);
        k_elu<<<cdiv((long)NG * HH, 256), 256, 0, stream>>>(ctx, (size_t)NG * HH);
        // GRU gates
        k_cvt_a16<<<cdiv((long)NG * KP_H, 256), 256, 0, stream>>>(ctx, c16, NG, HH, KP_H);
        k_cvt_w16t<<<cdiv((long)NQ_G * KP_H, 256), 256, 0, stream>>>(
            gru_Wih + (size_t)t * HH * 600, w16, HH, 600, KP_H, NQ_G);
        k_gemm_wmma<<<cdiv(NG / 16, 8), 256, 0, stream>>>(c16, w16, gru_bih + (size_t)t * 600,
                                                          gi, nullptr, NG, 600, KP_H, NQ_G, MODE_NONE);
        k_cvt_a16<<<cdiv((long)NG * KP_H, 256), 256, 0, stream>>>(gbuf, g16, NG, HH, KP_H);
        k_cvt_w16t<<<cdiv((long)NQ_G * KP_H, 256), 256, 0, stream>>>(
            gru_Whh + (size_t)t * HH * 600, w16, HH, 600, KP_H, NQ_G);
        k_gemm_wmma<<<cdiv(NG / 16, 8), 256, 0, stream>>>(g16, w16, gru_bhh + (size_t)t * 600,
                                                          gh, nullptr, NG, 600, KP_H, NQ_G, MODE_NONE);
        k_gru<<<cdiv((long)NG * HH, 256), 256, 0, stream>>>(gi, gh, gbuf);
    }

    // ---- output MLP ----
    k_cvt_a16<<<cdiv((long)NG * KP_H, 256), 256, 0, stream>>>(gbuf, g16, NG, HH, KP_H);
    k_cvt_w16t<<<cdiv((long)NQ_O1 * KP_H, 256), 256, 0, stream>>>(out1_W, w16, HH, 1024, KP_H, NQ_O1);
    k_gemm_wmma<<<cdiv(NG / 16, 8), 256, 0, stream>>>(g16, w16, out1_b, o1, nullptr,
                                                      NG, 1024, KP_H, NQ_O1, MODE_RELU);
    k_out2<<<NG, 256, 0, stream>>>(o1, out2_W, out2_b, out);
}
